// discriminator_24610162606746
// MI455X (gfx1250) — compile-verified, matrix operand order checked by reading, not run on hardware
//
#include <hip/hip_runtime.h>

typedef __attribute__((ext_vector_type(16))) _Float16 v16h;
typedef __attribute__((ext_vector_type(8)))  _Float16 v8h;
typedef __attribute__((ext_vector_type(8)))  float    v8f;
typedef __attribute__((ext_vector_type(4)))  unsigned int u32x4;
typedef __attribute__((ext_vector_type(8)))  unsigned int u32x8;

#define CHUNK_ 144
#define NKEY_  432
#define NKEYP_ 448

// ---------------------------------------------------------------------------
// Fragment gather: row-major [rows][32] f16 matrix in LDS -> WMMA fragment.
// A (16x32): lane m=lane&15, lanes 0-15 K=0..7,16..23; lanes 16-31 K=8..15,24..31
// B (32x16): identical pattern when the source is stored N-major (key rows,
// C contiguous), since B[k][n] = src[n][k].
// The two 8-half groups are 16B-contiguous -> 2x ds_load_b128.
// ---------------------------------------------------------------------------
__device__ inline v16h frag_rowmajor(const _Float16* base, int row0, int lane) {
  int r  = lane & 15;
  int kb = (lane < 16) ? 0 : 8;
  const _Float16* p = base + (row0 + r) * 32 + kb;
  v8h lo = *(const v8h*)(p);
  v8h hi = *(const v8h*)(p + 16);
  v16h f;
#pragma unroll
  for (int e = 0; e < 8; ++e) { f[e] = lo[e]; f[8 + e] = hi[e]; }
  return f;
}

// ---------------------------------------------------------------------------
// VALU-only 16-lane butterfly pieces via DPP16 (no LDS bpermute traffic):
// quad_perm xor1 (0xB1), quad_perm xor2 (0x4E), row_half_mirror (0x141),
// row_mirror (0x140). All stay within each 16-lane half of the wave.
// ---------------------------------------------------------------------------
template <int CTRL>
__device__ inline float dpp_movf(float v) {
  int r = __builtin_amdgcn_update_dpp(0, __float_as_int(v), CTRL, 0xF, 0xF, true);
  return __int_as_float(r);
}
__device__ inline float red16_max(float v) {
  v = fmaxf(v, dpp_movf<0xB1>(v));
  v = fmaxf(v, dpp_movf<0x4E>(v));
  v = fmaxf(v, dpp_movf<0x141>(v));
  v = fmaxf(v, dpp_movf<0x140>(v));
  return v;
}
__device__ inline float red16_sum(float v) {
  v += dpp_movf<0xB1>(v);
  v += dpp_movf<0x4E>(v);
  v += dpp_movf<0x141>(v);
  v += dpp_movf<0x140>(v);
  return v;
}

// ---------------------------------------------------------------------------
// CDNA5 async copy: 16 bytes global -> LDS, tracked by ASYNCcnt.
// ---------------------------------------------------------------------------
__device__ inline void async_copy16(unsigned int lds_off, const void* gaddr) {
  asm volatile("global_load_async_to_lds_b128 %0, %1, off"
               :: "v"(lds_off), "v"(gaddr) : "memory");
}
__device__ inline void wait_async0() {
  asm volatile("s_wait_asynccnt 0x0" ::: "memory");
}

// ---------------------------------------------------------------------------
// conv_in: 3x3 pad1, 1->32 channels, leaky_relu(0.1). fea layout [C][H][W]
// ---------------------------------------------------------------------------
__global__ void k_conv_in(const float* __restrict__ x, const float* __restrict__ w,
                          float* __restrict__ fea) {
  int t = blockIdx.x * blockDim.x + threadIdx.x;
  const int H = 192;
  if (t >= 32 * H * H) return;
  int co = t / (H * H);
  int yy = (t / H) % H;
  int xx = t % H;
  float acc = 0.f;
#pragma unroll
  for (int ky = 0; ky < 3; ++ky) {
#pragma unroll
    for (int kx = 0; kx < 3; ++kx) {
      int iy = yy + ky - 1, ix = xx + kx - 1;
      if (iy >= 0 && iy < H && ix >= 0 && ix < H)
        acc += x[iy * H + ix] * w[co * 9 + ky * 3 + kx];
    }
  }
  fea[t] = (acc > 0.f) ? acc : 0.1f * acc;
}

// ---------------------------------------------------------------------------
// conv_match: 3x3 pad1, Cin->Cout (+bias), output transposed as xe[L][Cout]
// ---------------------------------------------------------------------------
__global__ void k_conv_match(const float* __restrict__ fea, const float* __restrict__ wm,
                             const float* __restrict__ bm, float* __restrict__ xe,
                             int Cin, int Cout, int H) {
  int t = blockIdx.x * blockDim.x + threadIdx.x;
  int L = H * H;
  if (t >= L * Cout) return;
  int l = t / Cout, co = t % Cout;
  int yy = l / H, xx = l % H;
  float acc = bm[co];
  for (int ci = 0; ci < Cin; ++ci) {
    const float* fch = fea + (long)ci * L;
    const float* wch = wm + ((long)co * Cin + ci) * 9;
#pragma unroll
    for (int ky = 0; ky < 3; ++ky) {
#pragma unroll
      for (int kx = 0; kx < 3; ++kx) {
        int iy = yy + ky - 1, ix = xx + kx - 1;
        if (iy >= 0 && iy < H && ix >= 0 && ix < H)
          acc += fch[iy * H + ix] * wch[ky * 3 + kx];
      }
    }
  }
  xe[(long)l * Cout + co] = acc;
}

// ---------------------------------------------------------------------------
// conv_assembly: 1x1 Cin->Cin (+bias), output transposed yt[L][C]
// ---------------------------------------------------------------------------
__global__ void k_conv_assembly(const float* __restrict__ fea, const float* __restrict__ wa,
                                const float* __restrict__ ba, float* __restrict__ yt,
                                int C, int H) {
  int t = blockIdx.x * blockDim.x + threadIdx.x;
  int L = H * H;
  if (t >= L * C) return;
  int l = t / C, o = t % C;
  float acc = ba[o];
  for (int ci = 0; ci < C; ++ci)
    acc += fea[(long)ci * L + l] * wa[(long)o * C + ci];
  yt[(long)l * C + o] = acc;
}

// ---------------------------------------------------------------------------
// LSH: codes[h][l] = argmax over [rv, -rv] of x_embed . rot (first occurrence)
// rot layout [C][4][hb/2]. The rot[:,h,:] slice (strided 2D tile) is staged
// into LDS once per block via the Tensor Data Mover, then reused by all
// threads (reuse factor = block size).
// ---------------------------------------------------------------------------
__global__ void k_lsh(const float* __restrict__ xe, const float* __restrict__ rot,
                      int* __restrict__ codes, int C, int L, int hb) {
  __shared__ __align__(16) float srot[512];   // [C][hb2], C*hb2 <= 512
  const int hb2 = hb >> 1;
  const int h   = blockIdx.y;

  if (threadIdx.x == 0) {
    // D# group 0: count=1 | lds_addr | global_addr | type=2 ("image")
    unsigned long long ga = (unsigned long long)(const void*)(rot + (size_t)h * hb2);
    unsigned int la = (unsigned int)(unsigned long long)(void*)srot;
    u32x4 g0;
    g0[0] = 1u;
    g0[1] = la;
    g0[2] = (unsigned int)ga;
    g0[3] = (unsigned int)((ga >> 32) & 0x1FFFFFFull) | (2u << 30);
    // D# group 1: data_size=4B, tensor [hb2 x C], tile [hb2 x C],
    // dim0 stride = 4*hb2 elements (hash-major inner layout)
    unsigned long long st0 = (unsigned long long)(4u * (unsigned)hb2);
    unsigned int td0 = (unsigned)hb2, td1 = (unsigned)C;
    u32x8 g1;
    g1[0] = (2u << 16);                                   // data_size = 4 bytes
    g1[1] = (td0 & 0xFFFFu) << 16;                        // tensor_dim0 [15:0]
    g1[2] = ((td0 >> 16) & 0xFFFFu) | ((td1 & 0xFFFFu) << 16);
    g1[3] = ((td1 >> 16) & 0xFFFFu) | ((td0 & 0xFFFFu) << 16);  // tile_dim0
    g1[4] = (td1 & 0xFFFFu);                              // tile_dim1
    g1[5] = (unsigned int)(st0 & 0xFFFFFFFFull);          // dim0_stride lo
    g1[6] = (unsigned int)((st0 >> 32) & 0xFFFFull);      // dim0_stride hi
    g1[7] = 0u;
    asm volatile("tensor_load_to_lds %0, %1" :: "s"(g0), "s"(g1) : "memory");
    __builtin_amdgcn_s_wait_tensorcnt(0);
  }
  __syncthreads();

  int l = blockIdx.x * blockDim.x + threadIdx.x;
  if (l >= L) return;
  const float* xr = xe + (long)l * C;
  float xv[32];
#pragma unroll
  for (int c = 0; c < 32; ++c) xv[c] = (c < C) ? xr[c] : 0.f;
  float bp = -1e30f, bn = -1e30f;
  int ip = 0, inn = 0;
  for (int b = 0; b < hb2; ++b) {
    float r = 0.f;
    for (int c = 0; c < C; ++c) r += xv[c] * srot[c * hb2 + b];
    if (r > bp)  { bp = r;  ip = b; }
    if (-r > bn) { bn = -r; inn = b; }
  }
  codes[h * L + l] = (bn > bp) ? (hb2 + inn) : ip;
}

// ---------------------------------------------------------------------------
// Stable counting sort per hash round (codes < hb <= 128). One block per h.
// Phase 3 stages 4KB tiles of codes into LDS with async global->LDS copies
// so the per-bucket sequential scans hit LDS instead of global.
// ---------------------------------------------------------------------------
#define STILE 1024
__global__ void k_sort(const int* __restrict__ codes, int* __restrict__ idx,
                       int L, int hb) {
  __shared__ int cnt[128];
  __shared__ int base[128];
  __shared__ __align__(16) int tile[STILE];
  int h = blockIdx.x, tid = threadIdx.x;
  cnt[tid] = 0;
  __syncthreads();
  for (int l = tid; l < L; l += blockDim.x)
    atomicAdd(&cnt[codes[h * L + l]], 1);
  __syncthreads();
  if (tid == 0) {
    int s = 0;
    for (int b = 0; b < hb; ++b) { base[b] = s; s += cnt[b]; }
  }
  __syncthreads();

  int p = (tid < hb) ? base[tid] : 0;
  unsigned int tla = (unsigned int)(unsigned long long)(void*)tile;
  for (int t0 = 0; t0 < L; t0 += STILE) {
    int rem = (L - t0 < STILE) ? (L - t0) : STILE;
    for (int q = tid; q < (rem >> 2); q += blockDim.x)
      async_copy16(tla + (unsigned int)(q * 16), codes + (size_t)h * L + t0 + q * 4);
    wait_async0();
    __syncthreads();
    if (tid < hb) {
      for (int j = 0; j < rem; ++j)
        if (tile[j] == tid) idx[h * L + (p++)] = t0 + j;
    }
    __syncthreads();
  }
}

// ---------------------------------------------------------------------------
// Fused LSH attention: per (chunk k, hash h) block, 9 waves (one 16-row
// M-tile each). Flash-style online softmax; both GEMMs on v_wmma f16.
// Row max/sum reductions run on the VALU via DPP16 butterflies so the LDS
// pipe is dedicated to fragment feeds. NC = Cin/16. Output rows scatter
// through the sort permutation.
// ---------------------------------------------------------------------------
template <int NC>
__launch_bounds__(288)
__global__ void k_attn(const float* __restrict__ xe, const float* __restrict__ yt,
                       const int* __restrict__ idx, float* __restrict__ ret,
                       float* __restrict__ bs, int C, int Cin, int L, int nk) {
  __shared__ __align__(16) _Float16 sQ[CHUNK_ * 32];
  __shared__ __align__(16) _Float16 sK[NKEYP_ * 32];
  __shared__ int                    mrow[NKEYP_];
  __shared__ __align__(16) _Float16 pbuf[9 * 16 * 32];

  const int k   = blockIdx.x;
  const int h   = blockIdx.y;
  const int tid = threadIdx.x;
  const int km  = (k + nk - 1) % nk;
  const int kp  = (k + 1) % nk;

  // Cooperative gather + normalize into LDS (keys: [own | k-1 | k+1 | zero pad])
  for (int j = tid; j < NKEYP_; j += blockDim.x) {
    if (j < NKEY_) {
      int local = j % CHUNK_;
      int kc    = (j < CHUNK_) ? k : ((j < 2 * CHUNK_) ? km : kp);
      int g     = idx[h * L + kc * CHUNK_ + local];
      mrow[j]   = g;
      __builtin_prefetch(yt + (long)g * Cin, 0, 1);  // global_prefetch_b8
      const float* src = xe + (long)g * C;
      float v[32];
      float ss = 0.f;
#pragma unroll
      for (int c = 0; c < 32; ++c) {
        v[c] = (c < C) ? src[c] : 0.f;
        ss += v[c] * v[c];
      }
      float inv = 1.f / fmaxf(sqrtf(ss), 5e-5f);
#pragma unroll
      for (int c = 0; c < 32; ++c) {
        sK[j * 32 + c] = (_Float16)(v[c] * inv);
        if (j < CHUNK_) sQ[j * 32 + c] = (_Float16)v[c];
      }
    } else {
      mrow[j] = 0;
      for (int c = 0; c < 32; ++c) sK[j * 32 + c] = (_Float16)0.f;
    }
  }
  __syncthreads();

  const int wave = tid >> 5;   // 0..8 -> M-tile
  const int lane = tid & 31;
  const int half = lane >> 4;
  const int ln   = lane & 15;
  const int m0   = wave * 16;
  _Float16* pb   = pbuf + wave * 16 * 32;

  const v8f vzero = {0.f, 0.f, 0.f, 0.f, 0.f, 0.f, 0.f, 0.f};
  v8f acc[NC];
#pragma unroll
  for (int t = 0; t < NC; ++t) acc[t] = vzero;
  float mrun[8], lrun[8];
#pragma unroll
  for (int e = 0; e < 8; ++e) { mrun[e] = -1e30f; lrun[e] = 0.f; }

  const v16h aq = frag_rowmajor(sQ, m0, lane);

  for (int p = 0; p < 14; ++p) {   // 28 N-tiles (432 real keys + 16 pad), pairs of 2
    const int j0 = 2 * p, j1 = 2 * p + 1;
    v8f s0, s1;
    {
      v16h b0 = frag_rowmajor(sK, j0 * 16, lane);
      s0 = __builtin_amdgcn_wmma_f32_16x16x32_f16(false, aq, false, b0,
                                                  (short)0, vzero, false, false);
    }
    if (j1 < 27) {
      v16h b1 = frag_rowmajor(sK, j1 * 16, lane);
      s1 = __builtin_amdgcn_wmma_f32_16x16x32_f16(false, aq, false, b1,
                                                  (short)0, vzero, false, false);
    } else {
#pragma unroll
      for (int e = 0; e < 8; ++e) s1[e] = -1e30f;  // padded tile -> exp = 0
    }

#pragma unroll
    for (int e = 0; e < 8; ++e) {
      float hx   = red16_max(fmaxf(s0[e], s1[e]));      // row max within half
      float mnew = fmaxf(mrun[e], hx);
      float fac  = expf(mrun[e] - mnew);
      float p0   = expf(s0[e] - mnew);
      float p1   = expf(s1[e] - mnew);
      float rs   = red16_sum(p0 + p1);                  // row sum within half
      lrun[e] = lrun[e] * fac + rs;
      mrun[e] = mnew;
#pragma unroll
      for (int t = 0; t < NC; ++t) acc[t][e] = acc[t][e] * fac;
      int r = e + half * 8;                              // C-layout -> LDS stripe
      pb[r * 32 + ln]      = (_Float16)p0;
      pb[r * 32 + 16 + ln] = (_Float16)p1;
    }

    v16h pa = frag_rowmajor(pb, 0, lane);                // P as A-fragment [16x32]

    const int kb = (lane < 16) ? 0 : 8;
#pragma unroll
    for (int t = 0; t < NC; ++t) {
      v16h yb;
#pragma unroll
      for (int e = 0; e < 8; ++e) {
        int key0 = 32 * p + kb + e;
        int key1 = 32 * p + kb + 16 + e;
        yb[e]     = (_Float16)yt[(long)mrow[key0] * Cin + t * 16 + ln];
        yb[8 + e] = (_Float16)yt[(long)mrow[key1] * Cin + t * 16 + ln];
      }
      acc[t] = __builtin_amdgcn_wmma_f32_16x16x32_f16(false, pa, false, yb,
                                                      (short)0, acc[t], false, false);
    }
  }

  // Finalize: normalize, scatter through sort permutation, emit logsumexp.
#pragma unroll
  for (int e = 0; e < 8; ++e) {
    int r = m0 + e + half * 8;       // 0..143 (query rows == first 144 keys)
    int g = mrow[r];
    float invl = 1.f / lrun[e];
#pragma unroll
    for (int t = 0; t < NC; ++t)
      ret[((long)(h * L + g)) * Cin + t * 16 + ln] = acc[t][e] * invl;
    if (ln == 0)
      bs[h * L + g] = mrun[e] + logf(lrun[e]);
  }
}

// ---------------------------------------------------------------------------
// Combine: softmax over hash rounds, weighted sum, residual add.
// ---------------------------------------------------------------------------
__global__ void k_combine(const float* __restrict__ ret, const float* __restrict__ bs,
                          const float* __restrict__ fin, float* __restrict__ fout,
                          int C, int L) {
  int t = blockIdx.x * blockDim.x + threadIdx.x;
  if (t >= L * C) return;
  int ci = t / L, l = t % L;
  float b0 = bs[0 * L + l], b1 = bs[1 * L + l], b2 = bs[2 * L + l], b3 = bs[3 * L + l];
  float mx = fmaxf(fmaxf(b0, b1), fmaxf(b2, b3));
  float e0 = expf(b0 - mx), e1 = expf(b1 - mx), e2 = expf(b2 - mx), e3 = expf(b3 - mx);
  float inv = 1.f / (e0 + e1 + e2 + e3);
  float o = e0 * ret[((long)(0 * L + l)) * C + ci] + e1 * ret[((long)(1 * L + l)) * C + ci] +
            e2 * ret[((long)(2 * L + l)) * C + ci] + e3 * ret[((long)(3 * L + l)) * C + ci];
  fout[(long)ci * L + l] = o * inv + fin[(long)ci * L + l];
}

// ---------------------------------------------------------------------------
// Downsample: 4x4 stride-2 pad-1 conv, Cin -> 2*Cin.
// ---------------------------------------------------------------------------
__global__ void k_down(const float* __restrict__ fea, const float* __restrict__ w,
                       float* __restrict__ out, int Cin, int Cout, int H) {
  int t = blockIdx.x * blockDim.x + threadIdx.x;
  int Ho = H / 2;
  if (t >= Cout * Ho * Ho) return;
  int o  = t / (Ho * Ho);
  int oy = (t / Ho) % Ho;
  int ox = t % Ho;
  float acc = 0.f;
  for (int ci = 0; ci < Cin; ++ci) {
    const float* fch = fea + (long)ci * H * H;
    const float* wch = w + ((long)o * Cin + ci) * 16;
#pragma unroll
    for (int ky = 0; ky < 4; ++ky) {
#pragma unroll
      for (int kx = 0; kx < 4; ++kx) {
        int iy = oy * 2 + ky - 1, ix = ox * 2 + kx - 1;
        if (iy >= 0 && iy < H && ix >= 0 && ix < H)
          acc += fch[iy * H + ix] * wch[ky * 4 + kx];
      }
    }
  }
  out[t] = acc;
}

// ---------------------------------------------------------------------------
// Output conv: 3x3 valid, 256 -> 1, 24x24 -> 22x22.
// ---------------------------------------------------------------------------
__global__ void k_conv_out(const float* __restrict__ fea, const float* __restrict__ w,
                           float* __restrict__ out) {
  int t = blockIdx.x * blockDim.x + threadIdx.x;
  if (t >= 22 * 22) return;
  int oy = t / 22, ox = t % 22;
  float acc = 0.f;
  for (int ci = 0; ci < 256; ++ci) {
    const float* fch = fea + (long)ci * 24 * 24;
    const float* wch = w + (long)ci * 9;
#pragma unroll
    for (int ky = 0; ky < 3; ++ky)
#pragma unroll
      for (int kx = 0; kx < 3; ++kx)
        acc += fch[(oy + ky) * 24 + ox + kx] * wch[ky * 3 + kx];
  }
  out[t] = acc;
}

// ---------------------------------------------------------------------------
static inline int cdiv_(int a, int b) { return (a + b - 1) / b; }

extern "C" void kernel_launch(void* const* d_in, const int* in_sizes, int n_in,
                              void* d_out, int out_size, void* d_ws, size_t ws_size,
                              hipStream_t stream) {
  (void)in_sizes; (void)n_in; (void)out_size; (void)ws_size;
  static const int DIMS_[3] = {32, 64, 128};
  static const int HBS_[3]  = {128, 64, 16};

  const float* x   = (const float*)d_in[0];
  const float* win = (const float*)d_in[1];

  // Workspace partition (max sizes over stages; everything fits in L2)
  float* fa    = (float*)d_ws;
  float* fb    = fa + 1179648;          // fea ping/pong: 32*192*192
  float* xe    = fb + 1179648;          // x_embed [L][C]      <= 36864*8
  float* yt    = xe + 294912;           // y_embed [L][Cin]    <= 36864*32
  float* ret   = yt + 1179648;          // ret_uns [4][L][Cin] <= 4*36864*32
  float* bsb   = ret + 4718592;         // bucket logsumexp [4][L]
  int*   codes = (int*)(bsb + 147456);  // [4][L]
  int*   idxb  = codes + 147456;        // [4][L]

  float* cur = fa;
  float* nxt = fb;

  k_conv_in<<<cdiv_(32 * 192 * 192, 256), 256, 0, stream>>>(x, win, cur);

  int H = 192;
  for (int i = 0; i < 3; ++i) {
    const int d  = DIMS_[i];
    const int c  = d / 4;
    const int hb = HBS_[i];
    const int L  = H * H;
    const int nk = L / CHUNK_;

    for (int a = 0; a < 2; ++a) {
      const int pbase = 2 + i * 11 + a * 5;
      const float* wm  = (const float*)d_in[pbase + 0];
      const float* bm  = (const float*)d_in[pbase + 1];
      const float* wa  = (const float*)d_in[pbase + 2];
      const float* ba  = (const float*)d_in[pbase + 3];
      const float* rot = (const float*)d_in[pbase + 4];

      k_conv_match<<<cdiv_(L * c, 256), 256, 0, stream>>>(cur, wm, bm, xe, d, c, H);
      k_conv_assembly<<<cdiv_(L * d, 256), 256, 0, stream>>>(cur, wa, ba, yt, d, H);
      {
        dim3 gl(cdiv_(L, 256), 4);
        k_lsh<<<gl, 256, 0, stream>>>(xe, rot, codes, c, L, hb);
      }
      k_sort<<<4, 128, 0, stream>>>(codes, idxb, L, hb);

      dim3 g(nk, 4);
      if (d == 32)
        k_attn<2><<<g, 288, 0, stream>>>(xe, yt, idxb, ret, bsb, c, d, L, nk);
      else if (d == 64)
        k_attn<4><<<g, 288, 0, stream>>>(xe, yt, idxb, ret, bsb, c, d, L, nk);
      else
        k_attn<8><<<g, 288, 0, stream>>>(xe, yt, idxb, ret, bsb, c, d, L, nk);

      k_combine<<<cdiv_(L * d, 256), 256, 0, stream>>>(ret, bsb, cur, nxt, d, L);
      float* tmp = cur; cur = nxt; nxt = tmp;
    }

    const float* wd = (const float*)d_in[2 + i * 11 + 10];
    k_down<<<cdiv_(2 * d * (H / 2) * (H / 2), 256), 256, 0, stream>>>(cur, wd, nxt, d, 2 * d, H);
    float* tmp = cur; cur = nxt; nxt = tmp;
    H /= 2;
  }

  k_conv_out<<<cdiv_(484, 128), 128, 0, stream>>>(cur, (const float*)d_in[35], (float*)d_out);
}
